// GAT_4423816315101
// MI455X (gfx1250) — matrix operand order repaired and checked by reading, not compiled, not run on hardware
//
#include <hip/hip_runtime.h>
#include <hip/hip_bf16.h>
#include <math.h>

// ---------------- problem constants (match reference) ----------------
#define GN   50000      // nodes
#define GE   800000     // edges
#define GH   4          // heads
#define GD   64         // per-head dim
#define GC   256        // H*D == IN
#define NEG_SLOPE 0.2f

typedef __attribute__((ext_vector_type(16))) _Float16 v16h;
typedef __attribute__((ext_vector_type(8)))  _Float16 v8h;
typedef __attribute__((ext_vector_type(8)))  float    v8f;

union V16H { v16h v; v8h h[2]; };

// ---------------- helpers ----------------
__device__ __forceinline__ float atomicMaxF(float* addr, float val) {
    int old = __float_as_int(*addr);
    while (__int_as_float(old) < val) {
        int assumed = old;
        old = atomicCAS((int*)addr, assumed, __float_as_int(val));
        if (old == assumed) break;
    }
    return __int_as_float(old);
}

// ---------------- conversion kernels ----------------
__global__ void gat_f32_to_f16(const float* __restrict__ in, _Float16* __restrict__ out, int n) {
    int i = blockIdx.x * blockDim.x + threadIdx.x;
    if (i < n) out[i] = (_Float16)in[i];
}

// W[k][n] (row-major [256,256]) -> Wt[n][k] f16 (column-major for contiguous B-lane loads)
__global__ void gat_w_transpose_f16(const float* __restrict__ W, _Float16* __restrict__ Wt) {
    int i = blockIdx.x * blockDim.x + threadIdx.x;   // i over 256*256
    int k = i >> 8, n = i & 255;
    Wt[n * GC + k] = (_Float16)W[k * GC + n];
}

// ---------------- WMMA GEMM: C[M,256] = A[M,256] * W[256,256] ----------------
// A: f16 row-major. Bt: f16, Bt[n*256+k] = W[k][n]. C: f32 row-major.
// Block = 512 threads = 16 waves; each wave computes one 16x16 tile of one
// 16-row stripe; K loop = 8 x v_wmma_f32_16x16x32_f16.
__global__ __launch_bounds__(512)
void gat_gemm_wmma(const _Float16* __restrict__ A, const _Float16* __restrict__ Bt,
                   float* __restrict__ C, int M) {
    const int lane    = threadIdx.x & 31;
    const int wave    = threadIdx.x >> 5;          // 0..15 -> column tile
    const int rowBase = blockIdx.x << 4;
    const int laneHi  = lane >> 4;                 // 0 or 1
    const int l15     = lane & 15;
    const int row     = rowBase + l15;             // A row held by this lane
    const int col     = (wave << 4) + l15;         // B/C column held by this lane
    const _Float16* arow = A  + (size_t)row * GC;
    const _Float16* bcol = Bt + (size_t)col * GC;
    const int qa = laneHi * 8;                     // A K-offset within 16 (ISA 16-bit A layout)
    const int qb = laneHi * 8;                     // B K-offset within 16 (same layout as A)

    v8f acc = {};
#pragma unroll
    for (int k0 = 0; k0 < GC; k0 += 32) {
        V16H a, b;
        a.h[0] = *(const v8h*)(arow + k0 + qa);        // K = k0+qa .. +7
        a.h[1] = *(const v8h*)(arow + k0 + 16 + qa);   // K = k0+16+qa .. +7
        b.h[0] = *(const v8h*)(bcol + k0 + qb);        // K = k0+qb .. +7
        b.h[1] = *(const v8h*)(bcol + k0 + 16 + qb);   // K = k0+16+qb .. +7
        acc = __builtin_amdgcn_wmma_f32_16x16x32_f16(
                  false, a.v, false, b.v, (short)0, acc, false, false);
    }
    // D layout: VGPR r -> row (r + 8*laneHi), column col
    float* cp = C + (size_t)(rowBase + laneHi * 8) * GC + col;
#pragma unroll
    for (int r = 0; r < 8; ++r)
        cp[(size_t)r * GC] = acc[r];
}

// ---------------- attention scores: el/er [N,H] ----------------
// one wave32 per (node, head); 64-wide dot product, shuffle-reduce.
__global__ void gat_scores(const float* __restrict__ feat,
                           const float* __restrict__ al, const float* __restrict__ ar,
                           float* __restrict__ el, float* __restrict__ er) {
    int gw   = (blockIdx.x * blockDim.x + threadIdx.x) >> 5;
    int lane = threadIdx.x & 31;
    if (gw >= GN * GH) return;
    int n = gw >> 2, h = gw & 3;
    const float* f  = feat + (size_t)n * GC + h * GD;
    const float* a  = al + h * GD;
    const float* b  = ar + h * GD;
    float f0 = f[lane], f1 = f[lane + 32];
    float sl = f0 * a[lane] + f1 * a[lane + 32];
    float sr = f0 * b[lane] + f1 * b[lane + 32];
#pragma unroll
    for (int off = 16; off > 0; off >>= 1) {
        sl += __shfl_down(sl, off, 32);
        sr += __shfl_down(sr, off, 32);
    }
    if (lane == 0) { el[gw] = sl; er[gw] = sr; }
}

__global__ void gat_fill_f32(float* __restrict__ p, float v, int n) {
    int i = blockIdx.x * blockDim.x + threadIdx.x;
    if (i < n) p[i] = v;
}

// ---------------- edge softmax phase 1: per-dst max ----------------
__global__ void gat_edge_max(const int* __restrict__ src, const int* __restrict__ dst,
                             const float* __restrict__ el, const float* __restrict__ er,
                             float* __restrict__ emax) {
    int e = blockIdx.x * blockDim.x + threadIdx.x;
    if (e >= GE) return;
    int s = src[e], d = dst[e];
#pragma unroll
    for (int h = 0; h < GH; ++h) {
        float v = el[s * GH + h] + er[d * GH + h];
        v = v > 0.0f ? v : NEG_SLOPE * v;
        atomicMaxF(&emax[d * GH + h], v);
    }
}

// ---------------- edge softmax phase 2: exp + per-dst sum ----------------
__global__ void gat_edge_expsum(const int* __restrict__ src, const int* __restrict__ dst,
                                const float* __restrict__ el, const float* __restrict__ er,
                                const float* __restrict__ emax,
                                float* __restrict__ ee, float* __restrict__ denom) {
    int e = blockIdx.x * blockDim.x + threadIdx.x;
    if (e >= GE) return;
    int s = src[e], d = dst[e];
#pragma unroll
    for (int h = 0; h < GH; ++h) {
        float v = el[s * GH + h] + er[d * GH + h];
        v = v > 0.0f ? v : NEG_SLOPE * v;
        float ex = __expf(v - emax[d * GH + h]);
        ee[e * GH + h] = ex;
        atomicAdd(&denom[d * GH + h], ex);
    }
}

// ---------------- message + scatter aggregation ----------------
// one block (256 threads) per edge: coalesced read of feat[src], coalesced
// atomicAdd into agg[dst].
__global__ __launch_bounds__(GC)
void gat_edge_aggregate(const int* __restrict__ src, const int* __restrict__ dst,
                        const float* __restrict__ feat, const float* __restrict__ ee,
                        const float* __restrict__ denom, float* __restrict__ agg) {
    int e = blockIdx.x;
    int t = threadIdx.x;            // 0..255  (h = t/64, d = t%64)
    int h = t >> 6;
    int s = src[e], d = dst[e];
    float alpha = ee[e * GH + h] / denom[d * GH + h];
    atomicAdd(&agg[(size_t)d * GC + t], alpha * feat[(size_t)s * GC + t]);
}

// ---------------- finalize (layers 0,1): bias [+residual] + ELU, in place ----------------
__global__ void gat_finalize(float* __restrict__ agg, const float* __restrict__ h_in,
                             const float* __restrict__ bias, int residual, int activate, int n) {
    int i = blockIdx.x * blockDim.x + threadIdx.x;
    if (i >= n) return;
    float v = agg[i] + bias[i & (GC - 1)];
    if (residual) v += h_in[i];
    if (activate) v = v > 0.0f ? v : (__expf(v) - 1.0f);
    agg[i] = v;
}

// ---------------- finalize (layer 2): residual + bias, head-mean -> [N,64] ----------------
__global__ void gat_finalize_mean(const float* __restrict__ agg, const float* __restrict__ h_in,
                                  const float* __restrict__ bias, float* __restrict__ out) {
    int i = blockIdx.x * blockDim.x + threadIdx.x;   // over N*64
    if (i >= GN * GD) return;
    int n = i >> 6, dd = i & (GD - 1);
    float s = 0.0f;
#pragma unroll
    for (int h = 0; h < GH; ++h) {
        int c = h * GD + dd;
        s += agg[(size_t)n * GC + c] + h_in[(size_t)n * GC + c] + bias[c];
    }
    out[i] = s * 0.25f;
}

// ---------------- host-side layer driver ----------------
static void run_gat_layer(const float* h_in, const float* W, const float* al,
                          const float* ar, const float* bias,
                          const int* src, const int* dst,
                          _Float16* h16, _Float16* wt16, float* feat,
                          float* el, float* er, float* emax, float* denom, float* ee,
                          float* agg, int layer, float* mean_out, hipStream_t stream) {
    const int TPB = 256;
    gat_f32_to_f16<<<(GN * GC) / TPB, TPB, 0, stream>>>(h_in, h16, GN * GC);
    gat_w_transpose_f16<<<(GC * GC) / TPB, TPB, 0, stream>>>(W, wt16);
    gat_gemm_wmma<<<GN / 16, 512, 0, stream>>>(h16, wt16, feat, GN);
    gat_scores<<<(GN * GH * 32) / TPB, TPB, 0, stream>>>(feat, al, ar, el, er);
    gat_fill_f32<<<(GN * GH + TPB - 1) / TPB, TPB, 0, stream>>>(emax, -3.402823466e38f, GN * GH);
    gat_fill_f32<<<(GN * GH + TPB - 1) / TPB, TPB, 0, stream>>>(denom, 0.0f, GN * GH);
    gat_fill_f32<<<(GN * GC) / TPB, TPB, 0, stream>>>(agg, 0.0f, GN * GC);
    gat_edge_max<<<(GE + TPB - 1) / TPB, TPB, 0, stream>>>(src, dst, el, er, emax);
    gat_edge_expsum<<<(GE + TPB - 1) / TPB, TPB, 0, stream>>>(src, dst, el, er, emax, ee, denom);
    gat_edge_aggregate<<<GE, GC, 0, stream>>>(src, dst, feat, ee, denom, agg);
    if (layer < 2) {
        gat_finalize<<<(GN * GC) / TPB, TPB, 0, stream>>>(agg, h_in, bias,
                                                          layer > 0 ? 1 : 0, 1, GN * GC);
    } else {
        gat_finalize_mean<<<(GN * GD) / TPB, TPB, 0, stream>>>(agg, h_in, bias, mean_out);
    }
}

extern "C" void kernel_launch(void* const* d_in, const int* in_sizes, int n_in,
                              void* d_out, int out_size, void* d_ws, size_t ws_size,
                              hipStream_t stream) {
    (void)in_sizes; (void)n_in; (void)out_size; (void)ws_size;
    const float* x   = (const float*)d_in[0];
    const int*   src = (const int*)  d_in[1];
    const int*   dst = (const int*)  d_in[2];
    const float* W[3]  = { (const float*)d_in[3],  (const float*)d_in[7],  (const float*)d_in[11] };
    const float* al[3] = { (const float*)d_in[4],  (const float*)d_in[8],  (const float*)d_in[12] };
    const float* ar[3] = { (const float*)d_in[5],  (const float*)d_in[9],  (const float*)d_in[13] };
    const float* bs[3] = { (const float*)d_in[6],  (const float*)d_in[10], (const float*)d_in[14] };

    // workspace carve-up (all chunk sizes are multiples of 256 bytes)
    char* p = (char*)d_ws;
    const size_t SZ_NC  = (size_t)GN * GC * sizeof(float);     // 51.2 MB
    const size_t SZ_H16 = (size_t)GN * GC * sizeof(_Float16);  // 25.6 MB
    const size_t SZ_W16 = (size_t)GC * GC * sizeof(_Float16);  // 128 KB
    const size_t SZ_NH  = (size_t)GN * GH * sizeof(float);     // 800 KB
    const size_t SZ_EH  = (size_t)GE * GH * sizeof(float);     // 12.8 MB

    float*    buf0  = (float*)p;      p += SZ_NC;
    float*    buf1  = (float*)p;      p += SZ_NC;
    float*    feat  = (float*)p;      p += SZ_NC;
    _Float16* h16   = (_Float16*)p;   p += SZ_H16;
    _Float16* wt16  = (_Float16*)p;   p += SZ_W16;
    float*    el    = (float*)p;      p += SZ_NH;
    float*    er    = (float*)p;      p += SZ_NH;
    float*    emax  = (float*)p;      p += SZ_NH;
    float*    denom = (float*)p;      p += SZ_NH;
    float*    ee    = (float*)p;      p += SZ_EH;

    float* out = (float*)d_out;       // [N, 64] f32

    // layer 0: x -> buf0 ;  layer 1: buf0 -> buf1 ;  layer 2: buf1 -> d_out (agg in buf0)
    run_gat_layer(x,    W[0], al[0], ar[0], bs[0], src, dst, h16, wt16, feat,
                  el, er, emax, denom, ee, buf0, 0, nullptr, stream);
    run_gat_layer(buf0, W[1], al[1], ar[1], bs[1], src, dst, h16, wt16, feat,
                  el, er, emax, denom, ee, buf1, 1, nullptr, stream);
    run_gat_layer(buf1, W[2], al[2], ar[2], bs[2], src, dst, h16, wt16, feat,
                  el, er, emax, denom, ee, buf0, 2, out, stream);
}
